// SeqLabelRNN_25426206392923
// MI455X (gfx1250) — matrix operand order repaired and checked by reading, not compiled, not run on hardware
//
#include <hip/hip_runtime.h>

// ---------------------------------------------------------------------------
// SeqLabelRNN on gfx1250: f16 WMMA GEMMs + persistent weight-stationary
// recurrence kernel with software grid barriers.
// ---------------------------------------------------------------------------

typedef __attribute__((ext_vector_type(16))) _Float16 v16h;
typedef __attribute__((ext_vector_type(8)))  _Float16 v8h;
typedef __attribute__((ext_vector_type(8)))  float    v8f;

static constexpr int cB  = 128;
static constexpr int cT  = 512;
static constexpr int cE  = 256;
static constexpr int cH  = 512;
static constexpr int cVP = 128;
static constexpr int cVC = 64;

// ---------------------------------------------------------------------------
// WMMA helper: D = A(16x32 f16) * B(32x16 f16) + C(16x16 f32)
// ---------------------------------------------------------------------------
__device__ __forceinline__ v8f wmma_f16(v16h a, v16h b, v8f c) {
  return __builtin_amdgcn_wmma_f32_16x16x32_f16(
      /*neg_a=*/false, a, /*neg_b=*/false, b,
      /*c_mod=*/(short)0, c, /*reuse_a=*/false, /*reuse_b=*/false);
}

// A-fragment from a row-major f16 matrix (row pointer already includes M row).
// ISA 16-bit A layout: lane<16: elts 0..7 = K k0..k0+7, elts 8..15 = K k0+16..23
//                      lane>=16: elts 0..7 = K k0+8..15, elts 8..15 = K k0+24..31
__device__ __forceinline__ v16h load_afrag(const _Float16* __restrict__ rowptr,
                                           int kbase, int lane) {
  const int off = kbase + ((lane & 16) ? 8 : 0);
  v8h lo = *(const v8h*)(rowptr + off);
  v8h hi = *(const v8h*)(rowptr + off + 16);
  return __builtin_shufflevector(lo, hi, 0, 1, 2, 3, 4, 5, 6, 7,
                                 8, 9, 10, 11, 12, 13, 14, 15);
}

// B-fragment from the pre-packed weight buffer. Pack layout:
//   idx = (((ntile*KC + kc)*32 + lane)*16 + i),  value = W[k][n]
//   k = kc*32 + (lane>=16 ? 8 : 0) + (i<8 ? i : i+8),  n = ntile*16 + (lane&15)
__device__ __forceinline__ v16h load_bfrag(const _Float16* __restrict__ pack,
                                           int ntile, int kc, int KC, int lane) {
  return *(const v16h*)(pack + (((size_t)(ntile * KC + kc) * 32 + lane) * 16));
}

// ---------------------------------------------------------------------------
// Prep: f32->f16 embedding convert, weight packing, state/barrier zeroing.
// ---------------------------------------------------------------------------
__device__ __forceinline__ void pack_one(const float* __restrict__ W, int N,
                                         _Float16* __restrict__ dst, int o) {
  const int i    = o & 15;
  const int lane = (o >> 4) & 31;
  // rest = ntile*KC + kc, but the k/n math only needs (kc, ntile) jointly via:
  //   k = kc*32 + base + i ; n = ntile*16 + (lane&15)
  // dst index already encodes (ntile,kc) ordering; recover k,n from o:
  // o>>9 = ntile*KC + kc with KC = (K/32); caller passes K via N-agnostic trick:
  // we re-derive using the fact that pack is written linearly below.
  (void)W; (void)N; (void)dst; (void)i; (void)lane; // (specialized below)
}

__device__ __forceinline__ void pack_elem(const float* __restrict__ W, int N,
                                          int KC, _Float16* __restrict__ dst,
                                          int o) {
  const int i     = o & 15;
  const int lane  = (o >> 4) & 31;
  const int rest  = o >> 9;          // ntile*KC + kc
  const int kc    = rest % KC;
  const int ntile = rest / KC;
  const int k = kc * 32 + ((lane & 16) ? 8 : 0) + (i < 8 ? i : i + 8);
  const int n = ntile * 16 + (lane & 15);
  dst[o] = (_Float16)W[(size_t)k * N + n];
}

__global__ void prep_kernel(const float* __restrict__ emb_table,
                            const float* __restrict__ W_xh0,
                            const float* __restrict__ W_hh0,
                            const float* __restrict__ W_xh1,
                            const float* __restrict__ W_hh1,
                            const float* __restrict__ fc_w,
                            _Float16* __restrict__ emb16,
                            _Float16* __restrict__ pWxh0,
                            _Float16* __restrict__ pWhh0,
                            _Float16* __restrict__ pWxh1,
                            _Float16* __restrict__ pWhh1,
                            _Float16* __restrict__ pFcw,
                            float* __restrict__ h0_f32,
                            float* __restrict__ h1_f32,
                            _Float16* __restrict__ h0_f16,
                            _Float16* __restrict__ h1_f16,
                            unsigned* __restrict__ barrier_ct) {
  long o = (long)blockIdx.x * blockDim.x + threadIdx.x;
  const long EMB = (long)cVC * cE;   // 16384
  const long P0  = (long)cE * cH;    // 131072
  const long P1  = (long)cH * cH;    // 262144
  const long PF  = (long)cH * cVP;   // 65536
  const long ST  = (long)2 * cB * cH;// 131072 (both parity buffers)

  if (o < EMB) { emb16[o] = (_Float16)emb_table[o]; return; }
  o -= EMB;
  if (o < P0) { pack_elem(W_xh0, cH, cE / 32, pWxh0, (int)o); return; }
  o -= P0;
  if (o < P1) { pack_elem(W_hh0, cH, cH / 32, pWhh0, (int)o); return; }
  o -= P1;
  if (o < P1) { pack_elem(W_xh1, cH, cH / 32, pWxh1, (int)o); return; }
  o -= P1;
  if (o < P1) { pack_elem(W_hh1, cH, cH / 32, pWhh1, (int)o); return; }
  o -= P1;
  if (o < PF) { pack_elem(fc_w, cVP, cH / 32, pFcw, (int)o); return; }
  o -= PF;
  if (o < ST) {
    h0_f32[o] = 0.0f; h1_f32[o] = 0.0f;
    h0_f16[o] = (_Float16)0.0f; h1_f16[o] = (_Float16)0.0f;
    return;
  }
  o -= ST;
  if (o == 0) { *barrier_ct = 0u; }
}

// ---------------------------------------------------------------------------
// Kernel A: U0[(t,b),:] = emb16[src[b,t]] @ W_xh0 + b_xh0   (f16 out, (T,B,H))
// One block per 16-row M-tile (rows share t); 8 waves each own 4 N-tiles.
// ---------------------------------------------------------------------------
__global__ __launch_bounds__(256) void u0_gemm_kernel(
    const int* __restrict__ src, const _Float16* __restrict__ emb16,
    const _Float16* __restrict__ pWxh0, const float* __restrict__ b_xh0,
    _Float16* __restrict__ u0) {
  const int lane = threadIdx.x & 31;
  const int wv   = threadIdx.x >> 5;    // 0..7
  const int mt   = blockIdx.x;          // 0..4095
  const int t    = mt >> 3;
  const int b0   = (mt & 7) * 16;
  const int n0   = wv * 64;
  const int KC   = cE / 32;             // 8

  const int brow  = b0 + (lane & 15);
  const int vocab = src[(size_t)brow * cT + t];
  const _Float16* arow = emb16 + (size_t)vocab * cE;

  v8f acc0 = {}, acc1 = {}, acc2 = {}, acc3 = {};
#pragma unroll
  for (int kc = 0; kc < 8; ++kc) {
    v16h a = load_afrag(arow, kc * 32, lane);
    acc0 = wmma_f16(a, load_bfrag(pWxh0, (n0 >> 4) + 0, kc, KC, lane), acc0);
    acc1 = wmma_f16(a, load_bfrag(pWxh0, (n0 >> 4) + 1, kc, KC, lane), acc1);
    acc2 = wmma_f16(a, load_bfrag(pWxh0, (n0 >> 4) + 2, kc, KC, lane), acc2);
    acc3 = wmma_f16(a, load_bfrag(pWxh0, (n0 >> 4) + 3, kc, KC, lane), acc3);
  }
  v8f* accs[4] = {&acc0, &acc1, &acc2, &acc3};
#pragma unroll
  for (int j = 0; j < 4; ++j) {
    const int n = n0 + j * 16 + (lane & 15);
    const float bias = b_xh0[n];
#pragma unroll
    for (int v = 0; v < 8; ++v) {
      const int bb = b0 + ((lane & 16) ? 8 : 0) + v;
      u0[((size_t)t * cB + bb) * cH + n] = (_Float16)((*accs[j])[v] + bias);
    }
  }
}

// ---------------------------------------------------------------------------
// Grid barrier (monotonic counter; prep zeroes it each launch).
// ---------------------------------------------------------------------------
__device__ __forceinline__ void grid_barrier(unsigned* __restrict__ ct,
                                             unsigned barno) {
  __threadfence();          // make this thread's writes agent-visible
  __syncthreads();
  if (threadIdx.x == 0) {
    __hip_atomic_fetch_add(ct, 1u, __ATOMIC_RELEASE, __HIP_MEMORY_SCOPE_AGENT);
    const unsigned need = barno * gridDim.x;
    while (__hip_atomic_load(ct, __ATOMIC_ACQUIRE, __HIP_MEMORY_SCOPE_AGENT) <
           need)
      __builtin_amdgcn_s_sleep(1);
  }
  __syncthreads();
  __threadfence();          // acquire-side for the whole block
}

// ---------------------------------------------------------------------------
// Kernel B: persistent two-layer recurrence. 128 blocks x 64 threads =
// 256 waves; wave w owns output tile (mt = w>>5, nt = w&31). Recurrent
// weights for its N-strip live in VGPRs for all 512 steps (48 x v16h).
// ---------------------------------------------------------------------------
__global__ __launch_bounds__(64, 1) void rnn_recurrence_kernel(
    const _Float16* __restrict__ u0,        // (T,B,H) f16
    const _Float16* __restrict__ pWhh0, const _Float16* __restrict__ pWxh1,
    const _Float16* __restrict__ pWhh1, const float* __restrict__ b_xh1,
    const int* __restrict__ src_lengths,
    float* __restrict__ h0_f32, float* __restrict__ h1_f32,   // [2][B*H]
    _Float16* __restrict__ h0_f16, _Float16* __restrict__ h1_f16,
    _Float16* __restrict__ h1_all,          // (B,T,H) f16
    unsigned* __restrict__ barrier_ct) {
  const int lane = threadIdx.x & 31;
  const int gw   = blockIdx.x * 2 + (threadIdx.x >> 5);  // 0..255
  const int mt   = gw >> 5;                              // 0..7
  const int nt   = gw & 31;                              // 0..31
  const int m0   = mt * 16;
  const int KC   = cH / 32;                              // 16
  const int BH   = cB * cH;

  // Weight-stationary fragments (384 VGPRs).
  v16h whh0[16], wxh1[16], whh1[16];
#pragma unroll
  for (int kc = 0; kc < 16; ++kc) {
    whh0[kc] = load_bfrag(pWhh0, nt, kc, KC, lane);
    wxh1[kc] = load_bfrag(pWxh1, nt, kc, KC, lane);
    whh1[kc] = load_bfrag(pWhh1, nt, kc, KC, lane);
  }

  const int ncol      = nt * 16 + (lane & 15);
  const int mrow_base = m0 + ((lane & 16) ? 8 : 0);
  const int arow_m    = m0 + (lane & 15);

  int lens[8];
#pragma unroll
  for (int v = 0; v < 8; ++v) lens[v] = src_lengths[mrow_base + v];
  const float bias1 = b_xh1[ncol];

  unsigned barno = 0;
  for (int t = 0; t < cT; ++t) {
    const int p = t & 1, q = p ^ 1;
    if (t + 1 < cT)  // warm L2/L0 for next step's U0 slice
      __builtin_prefetch(u0 + ((size_t)(t + 1) * cB + mrow_base) * cH + ncol,
                         0, 1);

    // ---- layer 0: acc = U0[t] + h0_prev @ W_hh0 ----
    v8f acc;
#pragma unroll
    for (int v = 0; v < 8; ++v)
      acc[v] = (float)u0[((size_t)t * cB + (mrow_base + v)) * cH + ncol];
    {
      const _Float16* h0p16 = h0_f16 + q * BH + (size_t)arow_m * cH;
#pragma unroll
      for (int kc = 0; kc < 16; ++kc)
        acc = wmma_f16(load_afrag(h0p16, kc * 32, lane), whh0[kc], acc);
    }
    {
      float*    h0c   = h0_f32 + p * BH;
      const float* h0pr = h0_f32 + q * BH;
      _Float16* h0c16 = h0_f16 + p * BH;
#pragma unroll
      for (int v = 0; v < 8; ++v) {
        const int m = mrow_base + v;
        const float hp = h0pr[(size_t)m * cH + ncol];
        const float h  = (t < lens[v]) ? tanhf(acc[v]) : hp;
        h0c[(size_t)m * cH + ncol]   = h;
        h0c16[(size_t)m * cH + ncol] = (_Float16)h;
      }
    }
    grid_barrier(barrier_ct, ++barno);

    // ---- layer 1: acc = b1 + h0_cur @ W_xh1 + h1_prev @ W_hh1 ----
#pragma unroll
    for (int v = 0; v < 8; ++v) acc[v] = bias1;
    {
      const _Float16* h0c16 = h0_f16 + p * BH + (size_t)arow_m * cH;
#pragma unroll
      for (int kc = 0; kc < 16; ++kc)
        acc = wmma_f16(load_afrag(h0c16, kc * 32, lane), wxh1[kc], acc);
      const _Float16* h1p16 = h1_f16 + q * BH + (size_t)arow_m * cH;
#pragma unroll
      for (int kc = 0; kc < 16; ++kc)
        acc = wmma_f16(load_afrag(h1p16, kc * 32, lane), whh1[kc], acc);
    }
    {
      float*    h1c   = h1_f32 + p * BH;
      const float* h1pr = h1_f32 + q * BH;
      _Float16* h1c16 = h1_f16 + p * BH;
#pragma unroll
      for (int v = 0; v < 8; ++v) {
        const int m = mrow_base + v;
        const float hp = h1pr[(size_t)m * cH + ncol];
        const float h  = (t < lens[v]) ? tanhf(acc[v]) : hp;
        h1c[(size_t)m * cH + ncol]   = h;
        h1c16[(size_t)m * cH + ncol] = (_Float16)h;
        h1_all[((size_t)m * cT + t) * cH + ncol] = (_Float16)h;
      }
    }
    grid_barrier(barrier_ct, ++barno);
  }
}

// ---------------------------------------------------------------------------
// Kernel C: logits[(b,t),:] = h1_all[(b,t),:] @ fc_w + fc_b   (f32 out)
// One block per 16-row M-tile; 8 waves each own one of the 8 N-tiles.
// ---------------------------------------------------------------------------
__global__ __launch_bounds__(256) void logits_gemm_kernel(
    const _Float16* __restrict__ h1_all, const _Float16* __restrict__ pFcw,
    const float* __restrict__ fc_b, float* __restrict__ out) {
  const int lane = threadIdx.x & 31;
  const int wv   = threadIdx.x >> 5;   // n-tile 0..7
  const int r0   = blockIdx.x * 16;    // row tile base (rows = b*T + t)
  const int KC   = cH / 32;            // 16

  const _Float16* arow = h1_all + (size_t)(r0 + (lane & 15)) * cH;
  v8f acc = {};
#pragma unroll
  for (int kc = 0; kc < 16; ++kc)
    acc = wmma_f16(load_afrag(arow, kc * 32, lane),
                   load_bfrag(pFcw, wv, kc, KC, lane), acc);

  const int n = wv * 16 + (lane & 15);
  const float bias = fc_b[n];
#pragma unroll
  for (int v = 0; v < 8; ++v) {
    const int r = r0 + ((lane & 16) ? 8 : 0) + v;
    out[(size_t)r * cVP + n] = acc[v] + bias;
  }
}

// ---------------------------------------------------------------------------
// Host entry.
// ---------------------------------------------------------------------------
extern "C" void kernel_launch(void* const* d_in, const int* in_sizes, int n_in,
                              void* d_out, int out_size, void* d_ws,
                              size_t ws_size, hipStream_t stream) {
  (void)in_sizes; (void)n_in; (void)out_size; (void)ws_size;
  const int*   src         = (const int*)d_in[0];
  const int*   src_lengths = (const int*)d_in[1];
  const float* emb_table   = (const float*)d_in[2];
  const float* W_xh0       = (const float*)d_in[3];
  const float* b_xh0       = (const float*)d_in[4];
  const float* W_hh0       = (const float*)d_in[5];
  const float* W_xh1       = (const float*)d_in[6];
  const float* b_xh1       = (const float*)d_in[7];
  const float* W_hh1       = (const float*)d_in[8];
  const float* fc_w        = (const float*)d_in[9];
  const float* fc_b        = (const float*)d_in[10];

  char* ws = (char*)d_ws;
  size_t off = 0;
  auto alloc = [&](size_t bytes) -> void* {
    off = (off + 255) & ~(size_t)255;
    void* p = ws + off;
    off += bytes;
    return p;
  };

  _Float16* emb16  = (_Float16*)alloc((size_t)cVC * cE * 2);
  _Float16* pWxh0  = (_Float16*)alloc((size_t)cE * cH * 2);
  _Float16* pWhh0  = (_Float16*)alloc((size_t)cH * cH * 2);
  _Float16* pWxh1  = (_Float16*)alloc((size_t)cH * cH * 2);
  _Float16* pWhh1  = (_Float16*)alloc((size_t)cH * cH * 2);
  _Float16* pFcw   = (_Float16*)alloc((size_t)cH * cVP * 2);
  float*    h0_f32 = (float*)alloc((size_t)2 * cB * cH * 4);
  float*    h1_f32 = (float*)alloc((size_t)2 * cB * cH * 4);
  _Float16* h0_f16 = (_Float16*)alloc((size_t)2 * cB * cH * 2);
  _Float16* h1_f16 = (_Float16*)alloc((size_t)2 * cB * cH * 2);
  _Float16* u0     = (_Float16*)alloc((size_t)cT * cB * cH * 2);   // 64 MB
  _Float16* h1_all = (_Float16*)alloc((size_t)cB * cT * cH * 2);   // 64 MB
  unsigned* barrier_ct = (unsigned*)alloc(256);

  const long prep_total = (long)cVC * cE + (long)cE * cH + 3L * cH * cH +
                          (long)cH * cVP + 2L * cB * cH + 1;
  const int prep_blocks = (int)((prep_total + 255) / 256);
  prep_kernel<<<prep_blocks, 256, 0, stream>>>(
      emb_table, W_xh0, W_hh0, W_xh1, W_hh1, fc_w, emb16, pWxh0, pWhh0, pWxh1,
      pWhh1, pFcw, h0_f32, h1_f32, h0_f16, h1_f16, barrier_ct);

  u0_gemm_kernel<<<(cB * cT) / 16, 256, 0, stream>>>(src, emb16, pWxh0, b_xh0,
                                                     u0);

  rnn_recurrence_kernel<<<128, 64, 0, stream>>>(
      u0, pWhh0, pWxh1, pWhh1, b_xh1, src_lengths, h0_f32, h1_f32, h0_f16,
      h1_f16, h1_all, barrier_ct);

  logits_gemm_kernel<<<(cB * cT) / 16, 256, 0, stream>>>(h1_all, pFcw, fc_b,
                                                         (float*)d_out);
}
